// Attention_30803505447004
// MI455X (gfx1250) — matrix operand order repaired and checked by reading, not compiled, not run on hardware
//
#include <hip/hip_runtime.h>
#include <stdint.h>

// ---------------------------------------------------------------------------
// MI455X / gfx1250 attention layer.
// f16 WMMA (f32 accumulate) for all matmuls; async-to-LDS double buffering
// for the flash-attention K/V tiles.
// ---------------------------------------------------------------------------

typedef __attribute__((ext_vector_type(16))) _Float16 v16h;
typedef __attribute__((ext_vector_type(8)))  _Float16 v8h;
typedef __attribute__((ext_vector_type(8)))  float    v8f;

#define SEQ_N   1024
#define BATCH   4
#define CDIM    1024
#define NHEADS  16
#define HDIM    64
#define CACHE_L 1024
#define MTOT    (CACHE_L + SEQ_N)   // 2048
#define QKV_J   (3 * CDIM)          // 3072
#define MROWS   (SEQ_N * BATCH)     // 4096
#define QSCALE  0.125f              // 64^-0.5

// ---- WMMA helper ----------------------------------------------------------
__device__ __forceinline__ v8f wmma16(v16h a, v16h b, v8f c) {
  return __builtin_amdgcn_wmma_f32_16x16x32_f16(
      /*neg_a=*/false, a, /*neg_b=*/false, b,
      /*c_mod=*/(short)0, c, /*reuse_a=*/false, /*reuse_b=*/false);
}

__device__ __forceinline__ v16h cat16(v8h lo, v8h hi) {
  return __builtin_shufflevector(lo, hi, 0, 1, 2, 3, 4, 5, 6, 7,
                                         8, 9, 10, 11, 12, 13, 14, 15);
}

// A-fragment (16x32 f16): lane row = lane%16, K = 2*kb+e contiguous
// (16 elements), kb = 8*(lane/16).
__device__ __forceinline__ v16h load_a_f16(const _Float16* __restrict__ p,
                                           int kb) {
  v8h lo = *(const v8h*)(p + 2 * kb);
  v8h hi = *(const v8h*)(p + 2 * kb + 8);
  return cat16(lo, hi);
}

// B-fragment (32x16 f16): lane col = lane%16, K = 16*(lane/16)+e contiguous.
// Caller passes pointer already offset by (lane/16)*16 along K.
__device__ __forceinline__ v16h load_b_f16(const _Float16* __restrict__ p) {
  v8h lo = *(const v8h*)p;
  v8h hi = *(const v8h*)(p + 8);
  return cat16(lo, hi);
}

__device__ __forceinline__ v8f vzero8() {
  v8f z = {0.f, 0.f, 0.f, 0.f, 0.f, 0.f, 0.f, 0.f};
  return z;
}

// ---- CDNA5 async memory -> LDS (ASYNCcnt tracked) -------------------------
__device__ __forceinline__ void async_ld_b128(const void* lds_dst,
                                              const void* gsrc) {
  // Flat->LDS mapping keeps the AS3 byte offset in addr[31:0].
  unsigned lds = (unsigned)(uintptr_t)lds_dst;
  uint64_t ga = (uint64_t)(uintptr_t)gsrc;
  asm volatile("global_load_async_to_lds_b128 %0, %1, off"
               :: "v"(lds), "v"(ga)
               : "memory");
}
__device__ __forceinline__ void wait_async0() {
  asm volatile("s_wait_asynccnt 0" ::: "memory");
}

// ---------------------------------------------------------------------------
// K0a: elementwise fp32 -> f16
// ---------------------------------------------------------------------------
__global__ void convert_f32_f16(const float* __restrict__ src,
                                _Float16* __restrict__ dst, long n) {
  long i = (long)blockIdx.x * blockDim.x + threadIdx.x;
  if (i < n) dst[i] = (_Float16)src[i];
}

// ---------------------------------------------------------------------------
// K0b: weight transpose + fp32->f16:  Wt[j][k] = W[k][j]
// ---------------------------------------------------------------------------
__global__ void transpose_to_f16(const float* __restrict__ W,
                                 _Float16* __restrict__ Wt, int rowsK,
                                 int colsJ) {
  long idx = (long)blockIdx.x * blockDim.x + threadIdx.x;
  long total = (long)rowsK * colsJ;
  if (idx >= total) return;
  int j = (int)(idx / rowsK);
  int k = (int)(idx % rowsK);
  Wt[idx] = (_Float16)W[(long)k * colsJ + j];
}

// ---------------------------------------------------------------------------
// K1: pre_kv -> new_pre_kv rows [0,L) (fp32) + f16 K / V-transposed staging.
// ---------------------------------------------------------------------------
__global__ void copy_kv_kernel(const float* __restrict__ pre_kv,
                               float* __restrict__ out_kv,
                               _Float16* __restrict__ Kws,
                               _Float16* __restrict__ Vt) {
  size_t idx = (size_t)blockIdx.x * blockDim.x + threadIdx.x;
  const size_t total = (size_t)2 * BATCH * NHEADS * CACHE_L * HDIM;
  if (idx >= total) return;
  int d = (int)(idx % HDIM);
  size_t t = idx / HDIM;
  int l = (int)(t % CACHE_L); t /= CACHE_L;
  int h = (int)(t % NHEADS);  t /= NHEADS;
  int b = (int)(t % BATCH);
  int s = (int)(t / BATCH);
  float v = pre_kv[idx];
  out_kv[(((size_t)(s * BATCH + b) * NHEADS + h) * MTOT + l) * HDIM + d] = v;
  int bh = b * NHEADS + h;
  if (s == 0)
    Kws[((size_t)bh * MTOT + l) * HDIM + d] = (_Float16)v;
  else
    Vt[((size_t)bh * HDIM + d) * MTOT + l] = (_Float16)v;
}

// ---------------------------------------------------------------------------
// K2: QKV GEMM  (Xh[4096x1024] @ Wqkv[1024x3072] + b) with scatter epilogue.
// One wave computes a 32x64 output tile (2 A-frags, 8 wmma accumulators).
// ---------------------------------------------------------------------------
__global__ __launch_bounds__(256) void qkv_gemm_kernel(
    const _Float16* __restrict__ Xh, const _Float16* __restrict__ Wt,
    const float* __restrict__ bqkv, _Float16* __restrict__ Qws,
    _Float16* __restrict__ Kws, _Float16* __restrict__ Vt,
    float* __restrict__ out_kv) {
  const int lane = threadIdx.x & 31;
  const int wave = threadIdx.x >> 5;
  const int job = blockIdx.x * 8 + wave;         // 6144 jobs = 128 * 48
  const int m0 = (job / 48) * 32;
  const int n0 = (job % 48) * 64;
  const int col = lane & 15, hi = lane >> 4;
  const int kb = hi * 8;

  v8f acc[2][4];
#pragma unroll
  for (int r = 0; r < 2; ++r)
#pragma unroll
    for (int t = 0; t < 4; ++t) acc[r][t] = vzero8();

  const _Float16* ar0 = Xh + (size_t)(m0 + col) * CDIM;
  const _Float16* ar1 = ar0 + (size_t)16 * CDIM;
  for (int k0 = 0; k0 < CDIM; k0 += 32) {
    __builtin_prefetch(ar0 + k0 + 256, 0, 0);    // global_prefetch_b8
    v16h A0 = load_a_f16(ar0 + k0, kb);
    v16h A1 = load_a_f16(ar1 + k0, kb);
#pragma unroll
    for (int t = 0; t < 4; ++t) {
      const _Float16* brow =
          Wt + (size_t)(n0 + t * 16 + col) * CDIM + k0 + hi * 16;
      v16h B = load_b_f16(brow);
      acc[0][t] = wmma16(A0, B, acc[0][t]);
      acc[1][t] = wmma16(A1, B, acc[1][t]);
    }
  }

#pragma unroll
  for (int r = 0; r < 2; ++r) {
#pragma unroll
    for (int t = 0; t < 4; ++t) {
#pragma unroll
      for (int i = 0; i < 8; ++i) {
        int m = m0 + r * 16 + i + hi * 8;
        int j = n0 + t * 16 + col;
        float val = acc[r][t][i] + bqkv[j];
        int n = m >> 2, b = m & 3;               // m = n*B + b
        int which = j >> 10, h = (j >> 6) & 15, d = j & 63;
        int bh = b * NHEADS + h;
        if (which == 0) {
          Qws[((size_t)bh * SEQ_N + n) * HDIM + d] = (_Float16)(val * QSCALE);
        } else if (which == 1) {
          out_kv[(((size_t)b * NHEADS + h) * MTOT + CACHE_L + n) * HDIM + d] =
              val;
          Kws[((size_t)bh * MTOT + CACHE_L + n) * HDIM + d] = (_Float16)val;
        } else {
          out_kv[(((size_t)(BATCH + b) * NHEADS + h) * MTOT + CACHE_L + n) *
                     HDIM + d] = val;
          Vt[((size_t)bh * HDIM + d) * MTOT + CACHE_L + n] = (_Float16)val;
        }
      }
    }
  }
}

// ---------------------------------------------------------------------------
// K3: flash attention with async double-buffered K/V tiles in LDS.
// Grid (bh=64, nblk=16), 4 waves/block; waves share one (b,h), each owns a
// 16-row Q stripe.  Per 32-wide m-slab: 4 wmma QK^T + 4 wmma PV; P is
// transposed through LDS; K/V tiles staged with global_load_async_to_lds.
// ---------------------------------------------------------------------------
__global__ __launch_bounds__(128) void flash_attn_kernel(
    const _Float16* __restrict__ Qws, const _Float16* __restrict__ Kws,
    const _Float16* __restrict__ Vt, const int* __restrict__ mask,
    _Float16* __restrict__ AOh) {
  __shared__ __align__(16) _Float16 kbuf[2][32 * 64];  // [m-row][d]
  __shared__ __align__(16) _Float16 vbuf[2][64 * 32];  // [d][m-row]
  __shared__ __align__(16) _Float16 pbuf[4][16 * 40];  // per-wave P tile

  const int tid = threadIdx.x;
  const int lane = tid & 31;
  const int wave = tid >> 5;
  const int bh = blockIdx.x;                     // 0..63
  const int b = bh / NHEADS, h = bh % NHEADS;
  const int ntile = blockIdx.y * 64 + wave * 16;
  const int col = lane & 15, hi = lane >> 4;
  const int kb = hi * 8;

  const _Float16* Kbase = Kws + (size_t)bh * MTOT * HDIM;
  const _Float16* Vbase = Vt + (size_t)bh * HDIM * MTOT;

  const _Float16* qrow = Qws + ((size_t)bh * SEQ_N + ntile + col) * HDIM;
  v16h Q0 = load_a_f16(qrow, kb);                // head-dim K 0..31
  v16h Q1 = load_a_f16(qrow + 32, kb);           // head-dim K 32..63

  v8f o[4];
#pragma unroll
  for (int t = 0; t < 4; ++t) o[t] = vzero8();
  float rmax[8], rsum[8];
#pragma unroll
  for (int i = 0; i < 8; ++i) { rmax[i] = -1e30f; rsum[i] = 0.f; }

  // 128-thread cooperative staging: K tile 32x64 halves (256 b128 chunks),
  // V tile 64x32 halves (256 b128 chunks); thread covers chunks tid, tid+128.
  const int c1 = tid + 128;
  const int vr0 = tid >> 2, vo0 = (tid & 3) * 8;
  const int vr1 = c1 >> 2, vo1 = (c1 & 3) * 8;

  auto issue_tile = [&](int m0, int sel) {
    async_ld_b128(&kbuf[sel][tid * 8], Kbase + (size_t)m0 * HDIM + tid * 8);
    async_ld_b128(&kbuf[sel][c1 * 8], Kbase + (size_t)m0 * HDIM + c1 * 8);
    async_ld_b128(&vbuf[sel][tid * 8], Vbase + (size_t)vr0 * MTOT + m0 + vo0);
    async_ld_b128(&vbuf[sel][c1 * 8], Vbase + (size_t)vr1 * MTOT + m0 + vo1);
  };

  issue_tile(0, 0);

  for (int step = 0; step < MTOT / 32; ++step) {
    const int m0 = step * 32;
    const int sel = step & 1;
    wait_async0();                 // our async writes for buf[sel] done
    __syncthreads();               // everyone's writes done; prev reads done
    if (step + 1 < MTOT / 32) issue_tile(m0 + 32, sel ^ 1);

    const _Float16* kt = kbuf[sel];
    const _Float16* vtile = vbuf[sel];

    // --- S = Q @ K^T, two 16x16 column tiles (B-frags from LDS) -----------
    v8f sL = vzero8(), sR = vzero8();
    const _Float16* krowL = kt + (size_t)col * HDIM + hi * 16;
    const _Float16* krowR = kt + (size_t)(col + 16) * HDIM + hi * 16;
    sL = wmma16(Q0, load_b_f16(krowL), sL);
    sL = wmma16(Q1, load_b_f16(krowL + 32), sL);
    sR = wmma16(Q0, load_b_f16(krowR), sR);
    sR = wmma16(Q1, load_b_f16(krowR + 32), sR);

    // --- mask + online softmax over this 32-wide slab ---------------------
#pragma unroll
    for (int i = 0; i < 8; ++i) {
      int n = ntile + i + hi * 8;
      float sl = sL[i], sr = sR[i];
      if (mask[(size_t)n * MTOT + m0 + col]) sl = -INFINITY;
      if (mask[(size_t)n * MTOT + m0 + 16 + col]) sr = -INFINITY;
      float tmax = fmaxf(sl, sr);
#pragma unroll
      for (int off = 8; off; off >>= 1)
        tmax = fmaxf(tmax, __shfl_xor(tmax, off, 16));
      float mnew = fmaxf(rmax[i], tmax);         // >= -1e30, never -inf
      float alpha = __expf(rmax[i] - mnew);
      float pl = __expf(sl - mnew);
      float pr = __expf(sr - mnew);
      float ps = pl + pr;
#pragma unroll
      for (int off = 8; off; off >>= 1) ps += __shfl_xor(ps, off, 16);
      rsum[i] = rsum[i] * alpha + ps;
      rmax[i] = mnew;
      o[0][i] *= alpha; o[1][i] *= alpha; o[2][i] *= alpha; o[3][i] *= alpha;
      pbuf[wave][(i + hi * 8) * 40 + col] = (_Float16)pl;
      pbuf[wave][(i + hi * 8) * 40 + 16 + col] = (_Float16)pr;
    }

    // --- P(16x32) transposed via LDS into an A-fragment -------------------
    const _Float16* prow = &pbuf[wave][col * 40];
    v16h P = load_a_f16(prow, kb);

    // --- O += P @ V  (V-frags from LDS, contiguous along m) ---------------
#pragma unroll
    for (int t = 0; t < 4; ++t) {
      const _Float16* vptr = vtile + (size_t)(t * 16 + col) * 32 + hi * 16;
      o[t] = wmma16(P, load_b_f16(vptr), o[t]);
    }
  }

  // --- epilogue: O /= l, store f16 into AO[N,B,C] for the proj GEMM -------
#pragma unroll
  for (int t = 0; t < 4; ++t) {
#pragma unroll
    for (int i = 0; i < 8; ++i) {
      int n = ntile + i + hi * 8;
      float val = o[t][i] / rsum[i];
      AOh[((size_t)n * BATCH + b) * CDIM + h * HDIM + t * 16 + col] =
          (_Float16)val;
    }
  }
}

// ---------------------------------------------------------------------------
// K4: output projection  out = AO[4096x1024] @ Wproj[1024x1024] + b
// ---------------------------------------------------------------------------
__global__ __launch_bounds__(256) void proj_gemm_kernel(
    const _Float16* __restrict__ AOh, const _Float16* __restrict__ Wt,
    const float* __restrict__ bias, float* __restrict__ out) {
  const int lane = threadIdx.x & 31;
  const int wave = threadIdx.x >> 5;
  const int job = blockIdx.x * 8 + wave;         // 2048 jobs = 128 * 16
  const int m0 = (job / 16) * 32;
  const int n0 = (job % 16) * 64;
  const int col = lane & 15, hi = lane >> 4;
  const int kb = hi * 8;

  v8f acc[2][4];
#pragma unroll
  for (int r = 0; r < 2; ++r)
#pragma unroll
    for (int t = 0; t < 4; ++t) acc[r][t] = vzero8();

  const _Float16* ar0 = AOh + (size_t)(m0 + col) * CDIM;
  const _Float16* ar1 = ar0 + (size_t)16 * CDIM;
  for (int k0 = 0; k0 < CDIM; k0 += 32) {
    __builtin_prefetch(ar0 + k0 + 256, 0, 0);
    v16h A0 = load_a_f16(ar0 + k0, kb);
    v16h A1 = load_a_f16(ar1 + k0, kb);
#pragma unroll
    for (int t = 0; t < 4; ++t) {
      const _Float16* brow =
          Wt + (size_t)(n0 + t * 16 + col) * CDIM + k0 + hi * 16;
      v16h B = load_b_f16(brow);
      acc[0][t] = wmma16(A0, B, acc[0][t]);
      acc[1][t] = wmma16(A1, B, acc[1][t]);
    }
  }

#pragma unroll
  for (int r = 0; r < 2; ++r) {
#pragma unroll
    for (int t = 0; t < 4; ++t) {
#pragma unroll
      for (int i = 0; i < 8; ++i) {
        int m = m0 + r * 16 + i + hi * 8;
        int j = n0 + t * 16 + col;
        out[(size_t)m * CDIM + j] = acc[r][t][i] + bias[j];
      }
    }
  }
}

// ---------------------------------------------------------------------------
// Host launcher
// ---------------------------------------------------------------------------
extern "C" void kernel_launch(void* const* d_in, const int* in_sizes, int n_in,
                              void* d_out, int out_size, void* d_ws,
                              size_t ws_size, hipStream_t stream) {
  (void)in_sizes; (void)n_in; (void)out_size; (void)ws_size;
  const float* x      = (const float*)d_in[0];   // [N,B,C]
  const float* pre_kv = (const float*)d_in[1];   // [2,B,H,L,HD]
  const int*   mask   = (const int*)d_in[2];     // [N, L+N]
  const float* Wqkv   = (const float*)d_in[3];   // [C, 3C]
  const float* bqkv   = (const float*)d_in[4];   // [3C]
  const float* Wproj  = (const float*)d_in[5];   // [C, C]
  const float* bproj  = (const float*)d_in[6];   // [C]

  float* out    = (float*)d_out;                 // [N,B,C]
  float* out_kv = out + (size_t)MROWS * CDIM;    // [2,B,H,MTOT,HD]

  // Workspace carve-up (all f16 staging), ~64 MB total.
  _Float16* Xh      = (_Float16*)d_ws;
  _Float16* Qws     = Xh + (size_t)MROWS * CDIM;
  _Float16* Kws     = Qws + (size_t)BATCH * NHEADS * SEQ_N * HDIM;
  _Float16* Vtw     = Kws + (size_t)BATCH * NHEADS * MTOT * HDIM;
  _Float16* Wqkv_t  = Vtw + (size_t)BATCH * NHEADS * HDIM * MTOT;
  _Float16* Wproj_t = Wqkv_t + (size_t)QKV_J * CDIM;
  _Float16* AOh     = Wproj_t + (size_t)CDIM * CDIM;

  // K0: activation convert + weight transpose/convert
  convert_f32_f16<<<(MROWS * CDIM + 255) / 256, 256, 0, stream>>>(
      x, Xh, (long)MROWS * CDIM);
  transpose_to_f16<<<(QKV_J * CDIM + 255) / 256, 256, 0, stream>>>(
      Wqkv, Wqkv_t, CDIM, QKV_J);
  transpose_to_f16<<<(CDIM * CDIM + 255) / 256, 256, 0, stream>>>(
      Wproj, Wproj_t, CDIM, CDIM);

  // K1: KV-cache copy + staging
  copy_kv_kernel<<<(2 * BATCH * NHEADS * CACHE_L * HDIM) / 256, 256, 0,
                   stream>>>(pre_kv, out_kv, Kws, Vtw);

  // K2: QKV GEMM (6144 wave-jobs, 8 waves/block)
  qkv_gemm_kernel<<<768, 256, 0, stream>>>(Xh, Wqkv_t, bqkv, Qws, Kws, Vtw,
                                           out_kv);

  // K3: flash attention (async K/V staging)
  flash_attn_kernel<<<dim3(BATCH * NHEADS, SEQ_N / 64), 128, 0, stream>>>(
      Qws, Kws, Vtw, mask, AOh);

  // K4: output projection (2048 wave-jobs)
  proj_gemm_kernel<<<256, 256, 0, stream>>>(AOh, Wproj_t, bproj, out);
}
